// NewsEncoder_69922067579290
// MI455X (gfx1250) — matrix operand order repaired and checked by reading, not compiled, not run on hardware
//
#include <hip/hip_runtime.h>
#include <hip/hip_bf16.h>

typedef __attribute__((ext_vector_type(16))) _Float16 v16h;
typedef __attribute__((ext_vector_type(8)))  _Float16 v8h;
typedef __attribute__((ext_vector_type(8)))  float    v8f;
typedef int v4i __attribute__((vector_size(16)));
typedef __attribute__((address_space(1))) v4i as1_v4i;
typedef __attribute__((address_space(3))) v4i as3_v4i;

#define NEGF (-1e30f)

#define BN 512          // B*N
#define LT 32
#define LC 128
#define DREAL 300
#define DP 320          // padded D (mult of 32)
#define AREAL 200
#define AP 224          // padded A (mult of 32 for 2x2 tiling)
#define CDREAL 100
#define CDP 128         // padded CD (mult of 32)
#define NH 15
#define DK 20
#define HD 32           // padded per-head dim

#if __has_builtin(__builtin_amdgcn_global_load_async_to_lds_b128)
#define HAVE_ASYNC_LDS 1
#else
#define HAVE_ASYNC_LDS 0
#endif

union AFrag { v16h v; v8h h[2]; _Float16 e[16]; };

__device__ inline v8f wmma32(v16h a, v16h b, v8f c) {
  return __builtin_amdgcn_wmma_f32_16x16x32_f16(false, a, false, b, (short)0, c, false, false);
}

// A row-major [M, lda] f16; lane's A fragment for tile row m0, K offset k0.
__device__ inline v16h load_a_f16(const _Float16* A, int lda, int m0, int k0, int lane) {
  int hi = lane >> 4;
  const _Float16* row = A + (size_t)(m0 + (lane & 15)) * lda + k0 + hi * 16;
  AFrag f;
  f.h[0] = *(const v8h*)(row);
  f.h[1] = *(const v8h*)(row + 8);
  return f.v;
}

// B stored transposed: Bt [N, ldb] f16 (row n = column n of B). Lane's B fragment.
__device__ inline v16h load_b_f16(const _Float16* Bt, int ldb, int n0, int k0, int lane) {
  int hi = lane >> 4;
  const _Float16* row = Bt + (size_t)(n0 + (lane & 15)) * ldb + k0 + hi * 16;
  return *(const v16h*)(row);
}

// A fragment from f32 LDS (row-major [*, ld]) with convert-to-f16.
__device__ inline v16h load_a_lds_f32(const float* S, int ld, int m0, int k0, int lane) {
  int hi = lane >> 4;
  const float* row = S + (size_t)(m0 + (lane & 15)) * ld + k0 + hi * 16;
  AFrag f;
#pragma unroll
  for (int j = 0; j < 8; ++j) {
    f.e[j]     = (_Float16)row[j];
    f.e[8 + j] = (_Float16)row[8 + j];
  }
  return f.v;
}

// ---------------- cooperative B staging (32 x K stripe into LDS) ----------------

__device__ inline void wait_async0() {
#if HAVE_ASYNC_LDS
#if __has_builtin(__builtin_amdgcn_s_wait_asynccnt)
  __builtin_amdgcn_s_wait_asynccnt(0);
#else
  asm volatile("s_wait_asynccnt 0x0" ::: "memory");
#endif
#endif
}

// Bs: __shared__ [32*K]; Bt global [N, ldb]; stages rows n0..n0+31, K halves each.
__device__ inline void stage_b(_Float16* Bs, const _Float16* Bt, int ldb, int n0, int K) {
  int nchunks = 4 * K; // 32 rows * K halves / 8 halves-per-chunk
  for (int c = threadIdx.x; c < nchunks; c += 256) {
    int off = c * 8;
    int row = off / K, col = off % K;
    const _Float16* g = Bt + (size_t)(n0 + row) * ldb + col;
#if HAVE_ASYNC_LDS
    __builtin_amdgcn_global_load_async_to_lds_b128(
        (as1_v4i*)(void*)(const void*)g,
        (as3_v4i*)(void*)(Bs + off), 0, 0);
#else
    *(v8h*)(Bs + off) = *(const v8h*)g;
#endif
  }
  wait_async0();
  __syncthreads();
}

// B fragment out of staged LDS stripe (rows 0..31, ld = K).
__device__ inline v16h load_b_lds(const _Float16* Bs, int K, int n0, int k0, int lane) {
  int hi = lane >> 4;
  const _Float16* row = Bs + (size_t)(n0 + (lane & 15)) * K + k0 + hi * 16;
  return *(const v16h*)(row);
}

// 2x2 register-blocked tiles: 32x32 output per wave, B from LDS, A from global.
struct Acc4 { v8f a00, a01, a10, a11; };

__device__ inline Acc4 gemm_tile2x2(const _Float16* A, int lda, const _Float16* Bs,
                                    int K, int m0, int lane) {
  Acc4 t;
  t.a00 = {}; t.a01 = {}; t.a10 = {}; t.a11 = {};
  for (int k0 = 0; k0 < K; k0 += 32) {
    if (k0 + 64 < K)
      __builtin_prefetch(A + (size_t)(m0 + (lane & 15)) * lda + k0 + 64, 0, 0);
    v16h a0 = load_a_f16(A, lda, m0, k0, lane);
    v16h a1 = load_a_f16(A, lda, m0 + 16, k0, lane);
    v16h b0 = load_b_lds(Bs, K, 0, k0, lane);
    v16h b1 = load_b_lds(Bs, K, 16, k0, lane);
    t.a00 = wmma32(a0, b0, t.a00);
    t.a01 = wmma32(a0, b1, t.a01);
    t.a10 = wmma32(a1, b0, t.a10);
    t.a11 = wmma32(a1, b1, t.a11);
  }
  return t;
}

// ---------------- utility kernels ----------------

__global__ void zero_h(_Float16* p, long n) {
  long i = (long)blockIdx.x * blockDim.x + threadIdx.x;
  if (i < n) p[i] = (_Float16)0.f;
}

// W [Kreal, Nreal] f32 row-major -> Wt [Npad, Kpad] f16 (transposed, zero padded)
__global__ void conv_w(const float* W, int Kreal, int Nreal, _Float16* Wt, int Kpad, int Npad) {
  long i = (long)blockIdx.x * blockDim.x + threadIdx.x;
  if (i >= (long)Npad * Kpad) return;
  int n = (int)(i / Kpad), k = (int)(i % Kpad);
  float v = (n < Nreal && k < Kreal) ? W[(size_t)k * Nreal + n] : 0.f;
  Wt[(size_t)n * Kpad + k] = (_Float16)v;
}

__global__ void embed_words(const int* tok, const float* emb, _Float16* X, long rows) {
  long i = (long)blockIdx.x * blockDim.x + threadIdx.x;
  if (i >= rows * DP) return;
  long r = i / DP; int d = (int)(i % DP);
  float v = (d < DREAL) ? emb[(size_t)tok[r] * DREAL + d] : 0.f;
  X[i] = (_Float16)v;
}

__global__ void embed_cat(const int* idx, const float* emb, _Float16* E) {
  long i = (long)blockIdx.x * blockDim.x + threadIdx.x;
  if (i >= (long)BN * CDP) return;
  long r = i / CDP; int d = (int)(i % CDP);
  float v = (d < CDREAL) ? emb[(size_t)idx[r] * CDREAL + d] : 0.f;
  E[i] = (_Float16)v;
}

// ---------------- WMMA GEMM kernels ----------------
// Block = 256 threads = 8 waves; block covers 256 rows x 32 cols; B stripe staged in LDS.

// C f32 [M, ldc] = A[M,K] @ B[K,N] (+bias over Nreal).
__global__ __launch_bounds__(256) void gemm_f32(const _Float16* A, int lda,
                                                const _Float16* Bt, int ldb,
                                                const float* bias, int Nreal,
                                                float* C, int ldc, int M, int N, int K) {
  __shared__ __align__(32) _Float16 Bs[32 * DP];
  int lane = threadIdx.x & 31, w = threadIdx.x >> 5;
  int nstripes = N >> 5;
  int n0 = (blockIdx.x % nstripes) << 5;
  int m0 = ((blockIdx.x / nstripes) << 8) + (w << 5);
  stage_b(Bs, Bt, ldb, n0, K);
  Acc4 t = gemm_tile2x2(A, lda, Bs, K, m0, lane);
  const v8f* accs[4] = { &t.a00, &t.a01, &t.a10, &t.a11 };
#pragma unroll
  for (int mi = 0; mi < 2; ++mi)
#pragma unroll
    for (int ni = 0; ni < 2; ++ni) {
      const v8f& acc = *accs[mi * 2 + ni];
      int col = n0 + ni * 16 + (lane & 15);
      float b = (bias != nullptr && col < Nreal) ? bias[col] : 0.f;
      int rb = m0 + mi * 16 + ((lane >> 4) << 3);
#pragma unroll
      for (int r = 0; r < 8; ++r)
        C[(size_t)(rb + r) * ldc + col] = acc[r] + b;
    }
}

// QKV projection with per-head scatter. mode 0: dst[item,h,l,HD]; mode 1: dst[item,h,c,L].
__global__ __launch_bounds__(256) void gemm_qkv(const _Float16* A, const _Float16* Bt,
                                                const float* bias, _Float16* dst,
                                                int L, int mode, int M) {
  __shared__ __align__(32) _Float16 Bs[32 * DP];
  int lane = threadIdx.x & 31, w = threadIdx.x >> 5;
  int nstripes = DP >> 5;
  int n0 = (blockIdx.x % nstripes) << 5;
  int m0 = ((blockIdx.x / nstripes) << 8) + (w << 5);
  stage_b(Bs, Bt, DP, n0, DP);
  Acc4 t = gemm_tile2x2(A, DP, Bs, DP, m0, lane);
  const v8f* accs[4] = { &t.a00, &t.a01, &t.a10, &t.a11 };
#pragma unroll
  for (int mi = 0; mi < 2; ++mi)
#pragma unroll
    for (int ni = 0; ni < 2; ++ni) {
      const v8f& acc = *accs[mi * 2 + ni];
      int col = n0 + ni * 16 + (lane & 15);
      if (col >= DREAL) continue;
      int h = col / DK, c = col % DK;
      float b = bias[col];
      int rb = m0 + mi * 16 + ((lane >> 4) << 3);
#pragma unroll
      for (int r = 0; r < 8; ++r) {
        int row = rb + r;
        int item = row / L, l = row % L;
        _Float16 v = (_Float16)(acc[r] + b);
        if (mode == 0) dst[(((size_t)item * NH + h) * L + l) * HD + c] = v;
        else           dst[(((size_t)item * NH + h) * HD + c) * L + l] = v;
      }
    }
}

// Fused gate: outg = x * sigmoid(x@W + cross[item]); x f16 [M, DP].
__global__ __launch_bounds__(256) void gemm_gate(const _Float16* X, const _Float16* Bt,
                                                 const float* cross, _Float16* outg,
                                                 int L, int M) {
  __shared__ __align__(32) _Float16 Bs[32 * DP];
  int lane = threadIdx.x & 31, w = threadIdx.x >> 5;
  int nstripes = DP >> 5;
  int n0 = (blockIdx.x % nstripes) << 5;
  int m0 = ((blockIdx.x / nstripes) << 8) + (w << 5);
  stage_b(Bs, Bt, DP, n0, DP);
  Acc4 t = gemm_tile2x2(X, DP, Bs, DP, m0, lane);
  const v8f* accs[4] = { &t.a00, &t.a01, &t.a10, &t.a11 };
#pragma unroll
  for (int mi = 0; mi < 2; ++mi)
#pragma unroll
    for (int ni = 0; ni < 2; ++ni) {
      const v8f& acc = *accs[mi * 2 + ni];
      int col = n0 + ni * 16 + (lane & 15);
      int rb = m0 + mi * 16 + ((lane >> 4) << 3);
#pragma unroll
      for (int r = 0; r < 8; ++r) {
        int row = rb + r;
        int item = row / L;
        float g = acc[r] + cross[(size_t)item * DP + col];
        float s = 1.f / (1.f + __expf(-g));
        float x = (float)X[(size_t)row * DP + col];
        outg[(size_t)row * DP + col] = (_Float16)(x * s);
      }
    }
}

// ---------------- attention kernels ----------------

// Title: 4 waves/block, one (item,head) per wave. L=32.
__global__ __launch_bounds__(128) void attn_title(const _Float16* Qh, const _Float16* Kh,
                                                  const _Float16* Vt, const int* mask,
                                                  _Float16* out) {
  __shared__ float Sf[4][LT][LT];
  int lane = threadIdx.x & 31, w = threadIdx.x >> 5;
  int pair = blockIdx.x * 4 + w;
  int item = pair / NH, h = pair % NH;
  const _Float16* Q = Qh + (size_t)pair * LT * HD;
  const _Float16* K = Kh + (size_t)pair * LT * HD;
  const _Float16* V = Vt + (size_t)pair * HD * LT;
  const int* mrow = mask + item * LT;
  const float scale = 0.22360679775f; // 1/sqrt(20)
#pragma unroll
  for (int mi = 0; mi < 2; ++mi)
#pragma unroll
    for (int ni = 0; ni < 2; ++ni) {
      v8f acc = {};
      acc = wmma32(load_a_f16(Q, HD, mi * 16, 0, lane),
                   load_b_f16(K, HD, ni * 16, 0, lane), acc);
      int cc = ni * 16 + (lane & 15);
      bool ok = mrow[cc] > 0;
      int rb = mi * 16 + ((lane >> 4) << 3);
#pragma unroll
      for (int r = 0; r < 8; ++r)
        Sf[w][rb + r][cc] = ok ? acc[r] * scale : NEGF;
    }
  __syncthreads();
  {
    float* row = Sf[w][lane];
    float mx = row[0];
    for (int j = 1; j < LT; ++j) mx = fmaxf(mx, row[j]);
    float s = 0.f;
    for (int j = 0; j < LT; ++j) { float e = __expf(row[j] - mx); row[j] = e; s += e; }
    float inv = 1.f / s;
    for (int j = 0; j < LT; ++j) row[j] *= inv;
  }
  __syncthreads();
#pragma unroll
  for (int mi = 0; mi < 2; ++mi)
#pragma unroll
    for (int ni = 0; ni < 2; ++ni) {
      v8f acc = {};
      acc = wmma32(load_a_lds_f32(&Sf[w][0][0], LT, mi * 16, 0, lane),
                   load_b_f16(V, LT, ni * 16, 0, lane), acc);
      int c = ni * 16 + (lane & 15);
      if (c < DK) {
        int rb = mi * 16 + ((lane >> 4) << 3);
#pragma unroll
        for (int r = 0; r < 8; ++r)
          out[((size_t)item * LT + rb + r) * DP + h * DK + c] = (_Float16)acc[r];
      }
    }
}

// Content: 8 waves/block, one (item,head) per block, 16-row strip per wave. L=128.
__global__ __launch_bounds__(256) void attn_content(const _Float16* Qh, const _Float16* Kh,
                                                    const _Float16* Vt, const int* mask,
                                                    _Float16* out) {
  __shared__ float Sf[8][16][LC]; // 64 KB
  int lane = threadIdx.x & 31, w = threadIdx.x >> 5;
  int pair = blockIdx.x;
  int item = pair / NH, h = pair % NH;
  const _Float16* Q = Qh + (size_t)pair * LC * HD;
  const _Float16* K = Kh + (size_t)pair * LC * HD;
  const _Float16* V = Vt + (size_t)pair * HD * LC;
  const int* mrow = mask + item * LC;
  const float scale = 0.22360679775f;
  int m0 = w * 16;
#pragma unroll
  for (int nt = 0; nt < 8; ++nt) {
    v8f acc = {};
    acc = wmma32(load_a_f16(Q, HD, m0, 0, lane),
                 load_b_f16(K, HD, nt * 16, 0, lane), acc);
    int cc = nt * 16 + (lane & 15);
    bool ok = mrow[cc] > 0;
    int rb = (lane >> 4) << 3;
#pragma unroll
    for (int r = 0; r < 8; ++r)
      Sf[w][rb + r][cc] = ok ? acc[r] * scale : NEGF;
  }
  __syncthreads();
  if (lane < 16) {
    float* row = Sf[w][lane];
    float mx = row[0];
    for (int j = 1; j < LC; ++j) mx = fmaxf(mx, row[j]);
    float s = 0.f;
    for (int j = 0; j < LC; ++j) { float e = __expf(row[j] - mx); row[j] = e; s += e; }
    float inv = 1.f / s;
    for (int j = 0; j < LC; ++j) row[j] *= inv;
  }
  __syncthreads();
#pragma unroll
  for (int nt = 0; nt < 2; ++nt) {
    v8f acc = {};
    for (int k0 = 0; k0 < LC; k0 += 32)
      acc = wmma32(load_a_lds_f32(&Sf[w][0][0], LC, 0, k0, lane),
                   load_b_f16(V, LC, nt * 16, k0, lane), acc);
    int c = nt * 16 + (lane & 15);
    if (c < DK) {
      int rb = (lane >> 4) << 3;
#pragma unroll
      for (int r = 0; r < 8; ++r)
        out[((size_t)item * LC + m0 + rb + r) * DP + h * DK + c] = (_Float16)acc[r];
    }
  }
}

// ---------------- pooling kernels (VALU; cheap relative to GEMMs) ----------------

__global__ __launch_bounds__(256) void pool_add(const float* P, const float* v,
                                                const _Float16* x, const int* mask, int L,
                                                float* out, _Float16* out16) {
  __shared__ float al[LC];
  int item = blockIdx.x, t = threadIdx.x;
  for (int l = t; l < L; l += 256) {
    const float* p = P + ((size_t)item * L + l) * AP;
    float a = 0.f;
    for (int j = 0; j < AREAL; ++j) a += tanhf(p[j]) * v[j];
    al[l] = (mask[item * L + l] > 0) ? a : NEGF;
  }
  __syncthreads();
  if (t == 0) {
    float mx = al[0];
    for (int l = 1; l < L; ++l) mx = fmaxf(mx, al[l]);
    float s = 0.f;
    for (int l = 0; l < L; ++l) { float e = __expf(al[l] - mx); al[l] = e; s += e; }
    float inv = 1.f / s;
    for (int l = 0; l < L; ++l) al[l] *= inv;
  }
  __syncthreads();
  for (int d = t; d < DP; d += 256) {
    float acc = 0.f;
    if (d < DREAL)
      for (int l = 0; l < L; ++l) acc += al[l] * (float)x[((size_t)item * L + l) * DP + d];
    out[(size_t)item * DP + d] = acc;
    if (out16 != nullptr) out16[(size_t)item * DP + d] = (_Float16)acc;
  }
}

__global__ __launch_bounds__(256) void pool_cand(const float* P, const float* q,
                                                 const _Float16* x, const int* mask, int L,
                                                 float* out) {
  __shared__ float al[LC];
  int item = blockIdx.x, t = threadIdx.x;
  const float* qv = q + (size_t)item * AP;
  for (int l = t; l < L; l += 256) {
    const float* p = P + ((size_t)item * L + l) * AP;
    float a = 0.f;
    for (int j = 0; j < AREAL; ++j) a += p[j] * qv[j];
    a *= 0.070710678118654752f; // 1/sqrt(200)
    al[l] = (mask[item * L + l] > 0) ? a : NEGF;
  }
  __syncthreads();
  if (t == 0) {
    float mx = al[0];
    for (int l = 1; l < L; ++l) mx = fmaxf(mx, al[l]);
    float s = 0.f;
    for (int l = 0; l < L; ++l) { float e = __expf(al[l] - mx); al[l] = e; s += e; }
    float inv = 1.f / s;
    for (int l = 0; l < L; ++l) al[l] *= inv;
  }
  __syncthreads();
  for (int d = t; d < DP; d += 256) {
    float acc = 0.f;
    if (d < DREAL)
      for (int l = 0; l < L; ++l) acc += al[l] * (float)x[((size_t)item * L + l) * DP + d];
    out[(size_t)item * DP + d] = acc;
  }
}

__global__ void combine(const float* ts, const float* tc, const float* tsb,
                        const float* cs, const float* cc, const float* csb, float* out) {
  long i = (long)blockIdx.x * blockDim.x + threadIdx.x;
  if (i >= (long)BN * DREAL) return;
  int item = (int)(i / DREAL), d = (int)(i % DREAL);
  size_t si = (size_t)item * DP + d;
  out[(size_t)item * 2 * DREAL + d]         = ts[si] + tc[si] + tsb[si];
  out[(size_t)item * 2 * DREAL + DREAL + d] = cs[si] + cc[si] + csb[si];
}

// ---------------- host ----------------

static inline int cdiv(long a, long b) { return (int)((a + b - 1) / b); }

extern "C" void kernel_launch(void* const* d_in, const int* in_sizes, int n_in,
                              void* d_out, int out_size, void* d_ws, size_t ws_size,
                              hipStream_t stream) {
  (void)in_sizes; (void)n_in; (void)out_size; (void)ws_size;
  const int* title_text   = (const int*)d_in[0];
  const int* content_text = (const int*)d_in[1];
  const int* title_mask   = (const int*)d_in[2];
  const int* content_mask = (const int*)d_in[3];
  const int* category     = (const int*)d_in[4];
  const int* subCategory  = (const int*)d_in[5];
  const float* word_emb    = (const float*)d_in[6];
  const float* cate_emb    = (const float*)d_in[7];
  const float* subcate_emb = (const float*)d_in[8];
  const float* tWq = (const float*)d_in[9];  const float* tbq = (const float*)d_in[10];
  const float* tWk = (const float*)d_in[11]; const float* tbk = (const float*)d_in[12];
  const float* tWv = (const float*)d_in[13]; const float* tbv = (const float*)d_in[14];
  const float* cWq = (const float*)d_in[15]; const float* cbq = (const float*)d_in[16];
  const float* cWk = (const float*)d_in[17]; const float* cbk = (const float*)d_in[18];
  const float* cWv = (const float*)d_in[19]; const float* cbv = (const float*)d_in[20];
  const float* addt_W = (const float*)d_in[21]; const float* addt_b = (const float*)d_in[22];
  const float* addt_v = (const float*)d_in[23];
  const float* addc_W = (const float*)d_in[24]; const float* addc_b = (const float*)d_in[25];
  const float* addc_v = (const float*)d_in[26];
  const float* selft_W = (const float*)d_in[27]; const float* selft_b = (const float*)d_in[28];
  const float* selft_v = (const float*)d_in[29];
  const float* selfc_W = (const float*)d_in[30]; const float* selfc_b = (const float*)d_in[31];
  const float* selfc_v = (const float*)d_in[32];
  const float* cc_Wk = (const float*)d_in[33]; const float* cc_Wq = (const float*)d_in[34];
  const float* cc_bq = (const float*)d_in[35];
  const float* sc_Wk = (const float*)d_in[36]; const float* sc_Wq = (const float*)d_in[37];
  const float* sc_bq = (const float*)d_in[38];
  const float* tH = (const float*)d_in[39]; const float* tM = (const float*)d_in[40];
  const float* tM_b = (const float*)d_in[41];
  const float* cH = (const float*)d_in[42]; const float* cM = (const float*)d_in[43];
  const float* cM_b = (const float*)d_in[44];

  char* ws = (char*)d_ws;
  size_t off = 0;
  auto alloc = [&](size_t bytes) -> void* {
    off = (off + 255) & ~(size_t)255;
    void* p = ws + off; off += bytes; return p;
  };
  auto hbuf = [&](size_t n) { return (_Float16*)alloc(n * sizeof(_Float16)); };
  auto fbuf = [&](size_t n) { return (float*)alloc(n * sizeof(float)); };

  _Float16* tWqT = hbuf(DP * DP); _Float16* tWkT = hbuf(DP * DP); _Float16* tWvT = hbuf(DP * DP);
  _Float16* cWqT = hbuf(DP * DP); _Float16* cWkT = hbuf(DP * DP); _Float16* cWvT = hbuf(DP * DP);
  _Float16* tHT = hbuf(DP * DP); _Float16* tMT = hbuf(DP * DP);
  _Float16* cHT = hbuf(DP * DP); _Float16* cMT = hbuf(DP * DP);
  _Float16* addtWT = hbuf(AP * DP); _Float16* addcWT = hbuf(AP * DP);
  _Float16* selftWT = hbuf(AP * DP); _Float16* selfcWT = hbuf(AP * DP);
  _Float16* ccWkT = hbuf(AP * DP); _Float16* scWkT = hbuf(AP * DP);
  _Float16* ccWqT = hbuf(AP * CDP); _Float16* scWqT = hbuf(AP * CDP);

  _Float16* Xt = hbuf((size_t)BN * LT * DP);
  _Float16* Xc = hbuf((size_t)BN * LC * DP);
  _Float16* catE = hbuf((size_t)BN * CDP);
  _Float16* subE = hbuf((size_t)BN * CDP);

  size_t qkvN = (size_t)BN * NH * LC * HD; // content-sized, reused for title
  _Float16* Qh = hbuf(qkvN); _Float16* Kh = hbuf(qkvN); _Float16* Vt = hbuf(qkvN);

  _Float16* th  = hbuf((size_t)BN * LT * DP);
  _Float16* ch  = hbuf((size_t)BN * LC * DP);
  _Float16* thg = hbuf((size_t)BN * LT * DP);
  _Float16* chg = hbuf((size_t)BN * LC * DP);

  float* PA = fbuf((size_t)BN * LC * AP); // reused for every pooling projection
  float* crossT = fbuf((size_t)BN * DP);
  float* crossC = fbuf((size_t)BN * DP);
  float* qcat = fbuf((size_t)BN * AP);
  float* qsub = fbuf((size_t)BN * AP);
  float* tm = fbuf((size_t)BN * DP);   float* cm = fbuf((size_t)BN * DP);
  float* tself = fbuf((size_t)BN * DP); float* cself = fbuf((size_t)BN * DP);
  float* tcat = fbuf((size_t)BN * DP); float* tsub = fbuf((size_t)BN * DP);
  float* ccat = fbuf((size_t)BN * DP); float* csub = fbuf((size_t)BN * DP);
  _Float16* tm16 = hbuf((size_t)BN * DP);
  _Float16* cm16 = hbuf((size_t)BN * DP);

  // --- weight conversion ---
  int g320 = cdiv((long)DP * DP, 256), gA = cdiv((long)AP * DP, 256), gQ = cdiv((long)AP * CDP, 256);
  conv_w<<<g320, 256, 0, stream>>>(tWq, DREAL, DREAL, tWqT, DP, DP);
  conv_w<<<g320, 256, 0, stream>>>(tWk, DREAL, DREAL, tWkT, DP, DP);
  conv_w<<<g320, 256, 0, stream>>>(tWv, DREAL, DREAL, tWvT, DP, DP);
  conv_w<<<g320, 256, 0, stream>>>(cWq, DREAL, DREAL, cWqT, DP, DP);
  conv_w<<<g320, 256, 0, stream>>>(cWk, DREAL, DREAL, cWkT, DP, DP);
  conv_w<<<g320, 256, 0, stream>>>(cWv, DREAL, DREAL, cWvT, DP, DP);
  conv_w<<<g320, 256, 0, stream>>>(tH, DREAL, DREAL, tHT, DP, DP);
  conv_w<<<g320, 256, 0, stream>>>(tM, DREAL, DREAL, tMT, DP, DP);
  conv_w<<<g320, 256, 0, stream>>>(cH, DREAL, DREAL, cHT, DP, DP);
  conv_w<<<g320, 256, 0, stream>>>(cM, DREAL, DREAL, cMT, DP, DP);
  conv_w<<<gA, 256, 0, stream>>>(addt_W, DREAL, AREAL, addtWT, DP, AP);
  conv_w<<<gA, 256, 0, stream>>>(addc_W, DREAL, AREAL, addcWT, DP, AP);
  conv_w<<<gA, 256, 0, stream>>>(selft_W, DREAL, AREAL, selftWT, DP, AP);
  conv_w<<<gA, 256, 0, stream>>>(selfc_W, DREAL, AREAL, selfcWT, DP, AP);
  conv_w<<<gA, 256, 0, stream>>>(cc_Wk, DREAL, AREAL, ccWkT, DP, AP);
  conv_w<<<gA, 256, 0, stream>>>(sc_Wk, DREAL, AREAL, scWkT, DP, AP);
  conv_w<<<gQ, 256, 0, stream>>>(cc_Wq, CDREAL, AREAL, ccWqT, CDP, AP);
  conv_w<<<gQ, 256, 0, stream>>>(sc_Wq, CDREAL, AREAL, scWqT, CDP, AP);

  // --- embeddings ---
  embed_words<<<cdiv((long)BN * LT * DP, 256), 256, 0, stream>>>(title_text, word_emb, Xt, (long)BN * LT);
  embed_words<<<cdiv((long)BN * LC * DP, 256), 256, 0, stream>>>(content_text, word_emb, Xc, (long)BN * LC);
  embed_cat<<<cdiv((long)BN * CDP, 256), 256, 0, stream>>>(category, cate_emb, catE);
  embed_cat<<<cdiv((long)BN * CDP, 256), 256, 0, stream>>>(subCategory, subcate_emb, subE);

  // blocks = (M/256) * (N/32); all M here are multiples of 256, N multiples of 32.
  auto gemmf = [&](const _Float16* A, int lda, const _Float16* Bt, int ldb,
                   const float* bias, int Nreal, float* C, int ldc, int M, int N, int K) {
    gemm_f32<<<(M / 256) * (N / 32), 256, 0, stream>>>(A, lda, Bt, ldb, bias, Nreal, C, ldc, M, N, K);
  };

  // --- title branch MHSA ---
  long zt = (long)BN * NH * LT * HD;
  zero_h<<<cdiv(zt, 256), 256, 0, stream>>>(Qh, zt);
  zero_h<<<cdiv(zt, 256), 256, 0, stream>>>(Kh, zt);
  zero_h<<<cdiv(zt, 256), 256, 0, stream>>>(Vt, zt);
  {
    int M = BN * LT, blk = (M / 256) * (DP / 32);
    gemm_qkv<<<blk, 256, 0, stream>>>(Xt, tWqT, tbq, Qh, LT, 0, M);
    gemm_qkv<<<blk, 256, 0, stream>>>(Xt, tWkT, tbk, Kh, LT, 0, M);
    gemm_qkv<<<blk, 256, 0, stream>>>(Xt, tWvT, tbv, Vt, LT, 1, M);
  }
  attn_title<<<BN * NH / 4, 128, 0, stream>>>(Qh, Kh, Vt, title_mask, th);
  gemmf(th, DP, addtWT, DP, addt_b, AREAL, PA, AP, BN * LT, AP, DP);
  pool_add<<<BN, 256, 0, stream>>>(PA, addt_v, th, title_mask, LT, tm, tm16);

  // --- content branch MHSA ---
  long zc = (long)BN * NH * LC * HD;
  zero_h<<<cdiv(zc, 256), 256, 0, stream>>>(Qh, zc);
  zero_h<<<cdiv(zc, 256), 256, 0, stream>>>(Kh, zc);
  zero_h<<<cdiv(zc, 256), 256, 0, stream>>>(Vt, zc);
  {
    int M = BN * LC, blk = (M / 256) * (DP / 32);
    gemm_qkv<<<blk, 256, 0, stream>>>(Xc, cWqT, cbq, Qh, LC, 0, M);
    gemm_qkv<<<blk, 256, 0, stream>>>(Xc, cWkT, cbk, Kh, LC, 0, M);
    gemm_qkv<<<blk, 256, 0, stream>>>(Xc, cWvT, cbv, Vt, LC, 1, M);
  }
  attn_content<<<BN * NH, 256, 0, stream>>>(Qh, Kh, Vt, content_mask, ch);
  gemmf(ch, DP, addcWT, DP, addc_b, AREAL, PA, AP, BN * LC, AP, DP);
  pool_add<<<BN, 256, 0, stream>>>(PA, addc_v, ch, content_mask, LC, cm, cm16);

  // --- cross gating ---
  gemmf(cm16, DP, tMT, DP, tM_b, DREAL, crossT, DP, BN, DP, DP);
  gemmf(tm16, DP, cMT, DP, cM_b, DREAL, crossC, DP, BN, DP, DP);
  {
    int Mt = BN * LT;
    gemm_gate<<<(Mt / 256) * (DP / 32), 256, 0, stream>>>(th, tHT, crossT, thg, LT, Mt);
    int Mc = BN * LC;
    gemm_gate<<<(Mc / 256) * (DP / 32), 256, 0, stream>>>(ch, cHT, crossC, chg, LC, Mc);
  }

  // --- self pooling (on gated) ---
  gemmf(thg, DP, selftWT, DP, selft_b, AREAL, PA, AP, BN * LT, AP, DP);
  pool_add<<<BN, 256, 0, stream>>>(PA, selft_v, thg, title_mask, LT, tself, nullptr);
  gemmf(chg, DP, selfcWT, DP, selfc_b, AREAL, PA, AP, BN * LC, AP, DP);
  pool_add<<<BN, 256, 0, stream>>>(PA, selfc_v, chg, content_mask, LC, cself, nullptr);

  // --- candidate queries ---
  gemmf(catE, CDP, ccWqT, CDP, cc_bq, AREAL, qcat, AP, BN, AP, CDP);
  gemmf(subE, CDP, scWqT, CDP, sc_bq, AREAL, qsub, AP, BN, AP, CDP);

  // --- candidate attention pooling ---
  gemmf(thg, DP, ccWkT, DP, nullptr, AREAL, PA, AP, BN * LT, AP, DP);
  pool_cand<<<BN, 256, 0, stream>>>(PA, qcat, thg, title_mask, LT, tcat);
  gemmf(thg, DP, scWkT, DP, nullptr, AREAL, PA, AP, BN * LT, AP, DP);
  pool_cand<<<BN, 256, 0, stream>>>(PA, qsub, thg, title_mask, LT, tsub);
  gemmf(chg, DP, ccWkT, DP, nullptr, AREAL, PA, AP, BN * LC, AP, DP);
  pool_cand<<<BN, 256, 0, stream>>>(PA, qcat, chg, content_mask, LC, ccat);
  gemmf(chg, DP, scWkT, DP, nullptr, AREAL, PA, AP, BN * LC, AP, DP);
  pool_cand<<<BN, 256, 0, stream>>>(PA, qsub, chg, content_mask, LC, csub);

  // --- combine ---
  combine<<<cdiv((long)BN * DREAL, 256), 256, 0, stream>>>(tself, tcat, tsub, cself, ccat, csub,
                                                           (float*)d_out);
}
